// ARRNN_19679540150978
// MI455X (gfx1250) — compile-verified, matrix operand order checked by reading, not run on hardware
//
#include <hip/hip_runtime.h>

typedef __attribute__((ext_vector_type(16))) __bf16 v16bf;
typedef __attribute__((ext_vector_type(8)))  __bf16 v8bf;
typedef __attribute__((ext_vector_type(8)))  float  v8f;

#define B_   512
#define T_   1024
#define TM1  1023
#define I_   32
#define E_   32
#define H_   256
#define BT   32            // batch tile per workgroup
#define NWG  (B_/BT)       // 16 workgroups
#define K0   (I_+E_)       // 64
#define NTHREADS 512       // 16 waves

// ---- pre-swizzled bf16 weight-fragment layout in d_ws (bf16 element offsets) ----
#define FRAG_ELEMS 512     // one 32x16 (KxN) bf16 B-fragment = 32 lanes x 16 halves
#define OFF_IH0 0          // 48 ntiles x 2 ksteps
#define OFF_HH0 49152      // 48 x 8
#define OFF_IH1 245760     // 48 x 8
#define OFF_HH1 442368     // 48 x 8
#define OFF_OUT 638976     // 2  x 8
#define WS_ELEMS 647168

// ---- dynamic LDS layout (byte offsets) ----
#define SM_H0F   0          // fp32 h0 [32][256]
#define SM_H1F   32768      // fp32 h1 [32][256]
#define SM_Y     65536      // fp32 y  [32][32]
#define SM_H0A   69632      // bf16 h0 A-layout [32][264]
#define SM_H1A   86528      // bf16 h1 A-layout [32][264]
#define SM_INPA  103424     // bf16 inp A-layout [32][72]
#define SM_BIH0  108032     // fp32 biases (3H each)
#define SM_BHH0  111104
#define SM_BIH1  114176
#define SM_BHH1  117248
#define SM_BOUT  120320     // fp32 [32]
#define SM_BYTES 120448
#define HA_S  264           // bf16 row stride of h A-buffers (16B aligned, bank-skewed)
#define INP_S 72            // bf16 row stride of input A-buffer

typedef __attribute__((address_space(1))) const v16bf GBF;  // force GLOBAL path

// Scheduling-pipeline hints (composable-kernel style WMMA pipelining).
#if defined(__has_builtin) && __has_builtin(__builtin_amdgcn_sched_group_barrier)
#define SCHED_GROUP(mask, size, id) __builtin_amdgcn_sched_group_barrier(mask, size, id)
#else
#define SCHED_GROUP(mask, size, id)
#endif
#define SG_VMEM_READ 0x020
#define SG_DS_READ   0x100
#define SG_WMMA      0x008

__device__ __forceinline__ v8f wmma_bf(v16bf a, v16bf b, v8f c) {
  return __builtin_amdgcn_wmma_f32_16x16x32_bf16(false, a, false, b, (short)0, c,
                                                 false, false);
}

// Defeat LICM: make the weight stream pointer appear loop-variant each time step
// so B-fragment global loads stay inside the iteration (fresh L2 stream) instead
// of being hoisted into hundreds of VGPRs and spilled to scratch.
__device__ __forceinline__ const v16bf* launder(const v16bf* p) {
  asm volatile("" : "+s"(p));
  return p;
}

// A fragment (16x32 bf16, ISA layout): lanes 0-15 row M=lane, K {0..7,16..23};
// lanes 16-31 same rows, K {8..15,24..31}.  Two 16B LDS loads.
__device__ __forceinline__ v16bf ldsA(const __bf16* base, int stride, int mt,
                                      int ks, int lane) {
  int row = mt * 16 + (lane & 15);
  int k0  = ks * 32 + ((lane >= 16) ? 8 : 0);
  const __bf16* p = base + row * stride + k0;
  union { v16bf v; v8bf h[2]; } u;
  u.h[0] = *(const v8bf*)(p);        // K = k0 .. k0+7
  u.h[1] = *(const v8bf*)(p + 16);   // K = k0+16 .. k0+23
  return u.v;
}

// B fragment: 32 contiguous bytes per lane, explicit global address space so the
// backend emits global_load_b128 (LOADcnt only), never flat_load.
__device__ __forceinline__ v16bf gblB(const v16bf* w, int frag, int lane) {
  GBF* g = (GBF*)(w + frag * 32 + lane);
  return *g;
}

// Load the 6 B fragments (2 N-tiles x 3 gates) this wave needs for one k-step.
template <int KS>
__device__ __forceinline__ void loadB6(v16bf dst[6], const v16bf* w, int ks,
                                       int cg, int lane) {
#pragma unroll
  for (int nt = 0; nt < 2; ++nt)
#pragma unroll
    for (int g = 0; g < 3; ++g)
      dst[nt * 3 + g] = gblB(w, (g * 16 + cg * 2 + nt) * KS + ks, lane);
}

__device__ __forceinline__ float fsig(float x)  { return 1.f / (1.f + __expf(-x)); }
__device__ __forceinline__ float ftanh(float x) {
  float t = __expf(-2.f * fabsf(x));
  float r = (1.f - t) / (1.f + t);
  return (x >= 0.f) ? r : -r;
}

// One GRU layer update for this workgroup's 32-batch tile, 16 waves.
// Wave (mt,cg) owns rows [16mt,16mt+16) x hidden cols [32cg,32cg+32).
// B-fragment stream is double-buffered; sched_group_barrier pins the issue
// pattern per k-step: 12 global loads (next) -> 2 ds loads (cur A) -> 6 WMMAs
// (cur), so each load set has a full k-step of WMMA work to hide L2 latency.
// KSG is even (2 or 8) so buffer parity carries cleanly from gi into gh.
template <int KSG>
__device__ void layer_step(const __bf16* aGi, int sGi, const v16bf* wGi,
                           const __bf16* aGh, const v16bf* wGh,
                           float* hF, __bf16* hA,
                           const float* bih, const float* bhh,
                           int lane, int mt, int cg) {
  v8f accRZ[2][3];
  v8f accN[2];
#pragma unroll
  for (int nt = 0; nt < 2; ++nt) {
    accN[nt] = {};
#pragma unroll
    for (int g = 0; g < 3; ++g) accRZ[nt][g] = {};
  }

  v16bf bb[2][6];
  loadB6<KSG>(bb[0], wGi, 0, cg, lane);

  // gi = inp @ w_ih^T   (all three gates share accumulators with gh for r,z)
#pragma unroll
  for (int ks = 0; ks < KSG; ++ks) {
    int cur = ks & 1, nxt = cur ^ 1;
    if (ks + 1 < KSG) loadB6<KSG>(bb[nxt], wGi, ks + 1, cg, lane);
    else              loadB6<8>(bb[nxt], wGh, 0, cg, lane);   // bridge into gh
    v16bf a = ldsA(aGi, sGi, mt, ks, lane);
#pragma unroll
    for (int nt = 0; nt < 2; ++nt)
#pragma unroll
      for (int g = 0; g < 3; ++g)
        accRZ[nt][g] = wmma_bf(a, bb[cur][nt * 3 + g], accRZ[nt][g]);
    SCHED_GROUP(SG_VMEM_READ, 12, 0);
    SCHED_GROUP(SG_DS_READ,    2, 0);
    SCHED_GROUP(SG_WMMA,       6, 0);
  }

  // gh = h @ w_hh^T  (r,z join gi accumulators; n kept separate for r*h_n)
#pragma unroll
  for (int ks = 0; ks < 8; ++ks) {
    int cur = ks & 1, nxt = cur ^ 1;   // KSG even -> parity continues correctly
    if (ks + 1 < 8) loadB6<8>(bb[nxt], wGh, ks + 1, cg, lane);
    v16bf a = ldsA(aGh, HA_S, mt, ks, lane);
#pragma unroll
    for (int nt = 0; nt < 2; ++nt) {
      accRZ[nt][0] = wmma_bf(a, bb[cur][nt * 3 + 0], accRZ[nt][0]);
      accRZ[nt][1] = wmma_bf(a, bb[cur][nt * 3 + 1], accRZ[nt][1]);
      accN[nt]     = wmma_bf(a, bb[cur][nt * 3 + 2], accN[nt]);
    }
    if (ks + 1 < 8) { SCHED_GROUP(SG_VMEM_READ, 12, 0); }
    SCHED_GROUP(SG_DS_READ, 2, 0);
    SCHED_GROUP(SG_WMMA,    6, 0);
  }

  __syncthreads();   // all waves done reading hA before we overwrite it

#pragma unroll
  for (int nt = 0; nt < 2; ++nt) {
#pragma unroll
    for (int j = 0; j < 8; ++j) {
      int m   = mt * 16 + j + ((lane >= 16) ? 8 : 0);   // C/D VGPR layout
      int col = cg * 32 + nt * 16 + (lane & 15);
      float r = fsig(accRZ[nt][0][j] + bih[col]       + bhh[col]);
      float z = fsig(accRZ[nt][1][j] + bih[256 + col] + bhh[256 + col]);
      float n = ftanh(accRZ[nt][2][j] + bih[512 + col] +
                      r * (accN[nt][j] + bhh[512 + col]));
      float hold = hF[m * H_ + col];
      float hnew = (1.f - z) * n + z * hold;
      hF[m * H_ + col] = hnew;
      hA[m * HA_S + col] = (__bf16)hnew;
    }
  }
}

// ---- weight prep: fp32 -> bf16, swizzled into per-lane-contiguous B fragments ----
__global__ void grub_prep(const float* __restrict__ wih0, const float* __restrict__ whh0,
                          const float* __restrict__ wih1, const float* __restrict__ whh1,
                          const float* __restrict__ wout, __bf16* __restrict__ ws) {
  int gid = blockIdx.x * 256 + threadIdx.x;
  if (gid >= WS_ELEMS) return;
  const float* src; int kcols, ksteps, rem;
  if (gid < OFF_HH0)      { src = wih0; kcols = 64;  ksteps = 2; rem = gid - OFF_IH0; }
  else if (gid < OFF_IH1) { src = whh0; kcols = 256; ksteps = 8; rem = gid - OFF_HH0; }
  else if (gid < OFF_HH1) { src = wih1; kcols = 256; ksteps = 8; rem = gid - OFF_IH1; }
  else if (gid < OFF_OUT) { src = whh1; kcols = 256; ksteps = 8; rem = gid - OFF_HH1; }
  else                    { src = wout; kcols = 256; ksteps = 8; rem = gid - OFF_OUT; }
  int frag   = rem / FRAG_ELEMS, within = rem % FRAG_ELEMS;
  int lane   = within >> 4,      j      = within & 15;
  int ntile  = frag / ksteps,    kstep  = frag % ksteps;
  int n = ntile * 16 + (lane & 15);
  int k = kstep * 32 + ((j < 8) ? j : j + 8) + ((lane >= 16) ? 8 : 0);
  ws[gid] = (__bf16)src[n * kcols + k];
}

// ---- persistent GRU kernel: 16 workgroups x 32 batch, full time loop inside ----
__global__ __launch_bounds__(NTHREADS, 1) void grub_main(
    const float* __restrict__ X, const float* __restrict__ ET,
    const unsigned char* __restrict__ mask,
    const float* __restrict__ bih0, const float* __restrict__ bhh0,
    const float* __restrict__ bih1, const float* __restrict__ bhh1,
    const float* __restrict__ bout, const __bf16* __restrict__ ws,
    float* __restrict__ out) {
  extern __shared__ char sm[] __attribute__((aligned(16)));
  float*  h0f   = (float*)(sm + SM_H0F);
  float*  h1f   = (float*)(sm + SM_H1F);
  float*  yb    = (float*)(sm + SM_Y);
  __bf16* h0A   = (__bf16*)(sm + SM_H0A);
  __bf16* h1A   = (__bf16*)(sm + SM_H1A);
  __bf16* inpA  = (__bf16*)(sm + SM_INPA);
  float*  sbih0 = (float*)(sm + SM_BIH0);
  float*  sbhh0 = (float*)(sm + SM_BHH0);
  float*  sbih1 = (float*)(sm + SM_BIH1);
  float*  sbhh1 = (float*)(sm + SM_BHH1);
  float*  sbout = (float*)(sm + SM_BOUT);

  int tid = threadIdx.x, lane = tid & 31, wave = tid >> 5;   // 16 waves
  int mt = wave & 1, cg = wave >> 1;                         // cg: 0..7 (32 cols each)
  int wgBase = blockIdx.x * BT;

  for (int i = tid; i < BT * H_; i += NTHREADS)   { h0f[i] = 0.f; h1f[i] = 0.f; }
  for (int i = tid; i < BT * HA_S; i += NTHREADS) { h0A[i] = (__bf16)0.f; h1A[i] = (__bf16)0.f; }
  for (int i = tid; i < BT * I_; i += NTHREADS)   yb[i] = 0.f;
  for (int i = tid; i < 3 * H_; i += NTHREADS) {
    sbih0[i] = bih0[i]; sbhh0[i] = bhh0[i];
    sbih1[i] = bih1[i]; sbhh1[i] = bhh1[i];
  }
  if (tid < I_) sbout[tid] = bout[tid];
  __syncthreads();

  for (int t = 0; t < TM1; ++t) {
    // Re-launder weight streams each step: keeps B loads inside the iteration.
    const v16bf* wIH0 = launder((const v16bf*)(ws + OFF_IH0));
    const v16bf* wHH0 = launder((const v16bf*)(ws + OFF_HH0));
    const v16bf* wIH1 = launder((const v16bf*)(ws + OFF_IH1));
    const v16bf* wHH1 = launder((const v16bf*)(ws + OFF_HH1));
    const v16bf* wOUT = launder((const v16bf*)(ws + OFF_OUT));

    // Build inp = [ keep ? x_t : y_prev | e_{t+1} ] as bf16 A-tile.
    for (int idx = tid; idx < BT * K0; idx += NTHREADS) {
      int row = idx >> 6, col = idx & 63;
      int b = wgBase + row;
      float v;
      if (col < I_) {
        bool keep = (t == 0) || (mask[(size_t)b * T_ + t] != 0);
        v = keep ? X[((size_t)b * T_ + t) * I_ + col] : yb[row * I_ + col];
        if (col == 0 && t + 1 < TM1)
          __builtin_prefetch(&X[((size_t)b * T_ + t + 1) * I_], 0, 0);
      } else {
        v = ET[((size_t)b * T_ + t + 1) * E_ + (col - I_)];
      }
      inpA[row * INP_S + col] = (__bf16)v;
    }
    __syncthreads();

    layer_step<2>(inpA, INP_S, wIH0, h0A, wHH0, h0f, h0A, sbih0, sbhh0, lane, mt, cg);
    __syncthreads();
    layer_step<8>(h0A, HA_S, wIH1, h1A, wHH1, h1f, h1A, sbih1, sbhh1, lane, mt, cg);
    __syncthreads();

    // Readout y = h1 @ w_out^T + b_out : 2x2 tiles on waves 0..3.
    if (wave < 4) {
      int mtR = wave & 1, ntR = wave >> 1;
      v8f acc = {};
      v16bf b0 = gblB(wOUT, ntR * 8 + 0, lane);
#pragma unroll
      for (int ks = 0; ks < 8; ++ks) {
        v16bf bn;
        if (ks + 1 < 8) bn = gblB(wOUT, ntR * 8 + ks + 1, lane);
        v16bf a = ldsA(h1A, HA_S, mtR, ks, lane);
        acc = wmma_bf(a, b0, acc);
        b0 = bn;
      }
#pragma unroll
      for (int j = 0; j < 8; ++j) {
        int m   = mtR * 16 + j + ((lane >= 16) ? 8 : 0);
        int col = ntR * 16 + (lane & 15);
        float y = acc[j] + sbout[col];
        out[((size_t)(wgBase + m) * TM1 + t) * I_ + col] = y;
        yb[m * I_ + col] = y;
      }
    }
    __syncthreads();
  }

  // Final hidden state h1 -> tail of d_out.
  for (int i = tid; i < BT * H_; i += NTHREADS) {
    int row = i >> 8, col = i & 255;
    out[(size_t)B_ * TM1 * I_ + (size_t)(wgBase + row) * H_ + col] = h1f[i];
  }
}

extern "C" void kernel_launch(void* const* d_in, const int* in_sizes, int n_in,
                              void* d_out, int out_size, void* d_ws, size_t ws_size,
                              hipStream_t stream) {
  (void)in_sizes; (void)n_in; (void)out_size; (void)ws_size;
  const float* X    = (const float*)d_in[0];
  const float* ET   = (const float*)d_in[1];
  const unsigned char* mask = (const unsigned char*)d_in[2];
  const float* wih0 = (const float*)d_in[3];
  const float* whh0 = (const float*)d_in[4];
  const float* bih0 = (const float*)d_in[5];
  const float* bhh0 = (const float*)d_in[6];
  const float* wih1 = (const float*)d_in[7];
  const float* whh1 = (const float*)d_in[8];
  const float* bih1 = (const float*)d_in[9];
  const float* bhh1 = (const float*)d_in[10];
  const float* wout = (const float*)d_in[11];
  const float* bout = (const float*)d_in[12];
  __bf16* ws = (__bf16*)d_ws;
  float* out = (float*)d_out;

  grub_prep<<<(WS_ELEMS + 255) / 256, 256, 0, stream>>>(wih0, whh0, wih1, whh1, wout, ws);
  grub_main<<<NWG, NTHREADS, SM_BYTES, stream>>>(X, ET, mask, bih0, bhh0, bih1, bhh1,
                                                 bout, ws, out);
}
